// scGNN_26637387170502
// MI455X (gfx1250) — matrix operand order repaired and checked by reading, not compiled, not run on hardware
//
#include <hip/hip_runtime.h>

// ---------------------------------------------------------------------------
// CDNA5 (gfx1250) GCN forward: relu(GCN(x,W1,b1)) -> dropout -> GCN(.,W2,b2)
// GEMMs: v_wmma_f32_16x16x32_bf16, A band staged in LDS (bf16, padded,
// bank-conflict-free), 2 output tiles per wave for ILP on the matrix pipe.
// Layer-2 uses aggregate-then-GEMM (linearity) so both scatters are 256-dim.
// ---------------------------------------------------------------------------

typedef __bf16 bf16;
typedef __attribute__((ext_vector_type(16))) __bf16 v16bf;
typedef __attribute__((ext_vector_type(8)))  __bf16 v8bf;
typedef __attribute__((ext_vector_type(4)))  __bf16 v4bf;
typedef __attribute__((ext_vector_type(8)))  float  v8f;
typedef __attribute__((ext_vector_type(4)))  float  v4f;

#define N_NODES 50000
#define N_EDGES 800000
#define F_IN    1024
#define F_MID   256

// ------------------------------- utilities ---------------------------------

__global__ void fill_f32(float* __restrict__ p, float v, int n) {
    int i = blockIdx.x * blockDim.x + threadIdx.x;
    if (i < n) p[i] = v;
}

__global__ void degree_kernel(const long long* __restrict__ dst, int nE,
                              float* __restrict__ deg) {
    int i = blockIdx.x * blockDim.x + threadIdx.x;
    if (i < nE) atomicAdd(&deg[dst[i]], 1.0f);
}

__global__ void rsqrt_kernel(const float* __restrict__ deg,
                             float* __restrict__ dinv, int n) {
    int i = blockIdx.x * blockDim.x + threadIdx.x;
    if (i < n) {
        float d = deg[i];
        dinv[i] = (d > 0.0f) ? __frsqrt_rn(d) : 0.0f;
    }
}

// W  : [K][N] f32 row-major  ->  Wt : [N][K] bf16 row-major
__global__ void transpose_to_bf16(const float* __restrict__ W,
                                  bf16* __restrict__ Wt, int K, int N) {
    int o = blockIdx.x * blockDim.x + threadIdx.x;   // o = n*K + k
    if (o < K * N) {
        int n = o / K;
        int k = o - n * K;
        Wt[o] = (bf16)W[(size_t)k * N + n];
    }
}

// ------------------------------ WMMA GEMM ----------------------------------
// C[M][N] = A[M][K] (f32) x Bt[N][K] (bf16) (+bias)
// grid.x = (M/16) * blocksPerRow, 256 threads = 8 waves per block.
// Block: one 16-row A band (staged f32->bf16 in LDS once) x 256 columns.
// Wave w computes col-tiles (tnBase + 2w) and (tnBase + 2w + 1), sharing one
// A fragment per k-step across two independent WMMA accumulator chains.
__global__ void gemm_wmma_bf16(const float* __restrict__ A,
                               const bf16*  __restrict__ Bt,
                               const float* __restrict__ bias,
                               float* __restrict__ C,
                               int N, int K, int blocksPerRow) {
    extern __shared__ bf16 sA[];                     // [16][K+8] bf16
    const int KP = K + 8;                            // pad: row stride 2064 B
                                                     // -> 4-bank skew, no LDS
                                                     // conflicts on b128 reads
    const int tm     = blockIdx.x / blocksPerRow;
    const int tnBase = (blockIdx.x - tm * blocksPerRow) * 16;   // in 16-tiles

    // ---- phase 1: cooperative A-band load, f32 -> bf16 -> LDS
    const float* Abase  = A + (size_t)tm * 16 * K;
    const int    chunks = 16 * (K / 4);              // float4 chunks
    for (int c = threadIdx.x; c < chunks; c += 256) {
        int r  = c / (K / 4);
        int kq = (c - r * (K / 4)) * 4;
        v4f v  = *(const v4f*)(Abase + (size_t)r * K + kq);
        v4bf p;
        p[0] = (bf16)v[0]; p[1] = (bf16)v[1];
        p[2] = (bf16)v[2]; p[3] = (bf16)v[3];
        *(v4bf*)(sA + r * KP + kq) = p;              // ds_store_b64
    }
    __syncthreads();

    // ---- phase 2: WMMA over K
    const int lane = threadIdx.x & 31;
    const int wave = threadIdx.x >> 5;
    const int l15  = lane & 15;
    const int hi   = lane >> 4;                      // lane group 0/1
    const int col0 = (tnBase + wave * 2) * 16 + l15; // B col / C col, tile 0
    const int col1 = col0 + 16;                      //                tile 1

    const bf16* arow  = sA + l15 * KP;
    const bf16* b0row = Bt + (size_t)col0 * K;
    const bf16* b1row = Bt + (size_t)col1 * K;

    v8f acc0 = {}, acc1 = {};
    for (int k = 0; k < K; k += 32) {
        // A fragment (16-bit A 16x32): lanes<16 K=k..k+7,k+16..23 ;
        //                              lanes>=16 K=k+8..15,k+24..31
        const int kb = k + (hi ? 8 : 0);
        v8bf alo = *(const v8bf*)(arow + kb);        // ds_load_b128
        v8bf ahi = *(const v8bf*)(arow + kb + 16);   // ds_load_b128
        v16bf af;
#pragma unroll
        for (int i = 0; i < 8; ++i) { af[i] = alo[i]; af[i + 8] = ahi[i]; }

        // B fragments: lanes 0-15 K=k..k+7,k+16..23 ; lanes 16-31 K=k+8..15,k+24..31
        const int kb2 = k + (hi ? 8 : 0);
        v8bf p0 = *(const v8bf*)(b0row + kb2);
        v8bf p1 = *(const v8bf*)(b0row + kb2 + 16);
        v8bf q0 = *(const v8bf*)(b1row + kb2);
        v8bf q1 = *(const v8bf*)(b1row + kb2 + 16);
        // pull next k-slab of B toward the WGP while WMMA runs
        __builtin_prefetch(b0row + k + 32, 0, 1);    // global_prefetch_b8
        __builtin_prefetch(b1row + k + 32, 0, 1);
        v16bf bf0, bf1;
#pragma unroll
        for (int i = 0; i < 8; ++i) {
            bf0[i] = p0[i]; bf0[i + 8] = p1[i];
            bf1[i] = q0[i]; bf1[i + 8] = q1[i];
        }

        acc0 = __builtin_amdgcn_wmma_f32_16x16x32_bf16(
            false, af, false, bf0, (short)0, acc0, false, false);
        acc1 = __builtin_amdgcn_wmma_f32_16x16x32_bf16(
            false, af, false, bf1, (short)0, acc1, false, false);
    }

    // ---- store: VGPR i -> row tm*16 + i + (hi?8:0), col fixed per lane
    const float bv0 = bias ? bias[col0] : 0.0f;
    const float bv1 = bias ? bias[col1] : 0.0f;
    const int   r0  = tm * 16 + (hi ? 8 : 0);
#pragma unroll
    for (int i = 0; i < 8; ++i) {
        C[(size_t)(r0 + i) * N + col0] = acc0[i] + bv0;
        C[(size_t)(r0 + i) * N + col1] = acc1[i] + bv1;
    }
}

// ------------------------------- scatter -----------------------------------
// out[dst] += dinv[src]*dinv[dst] * H[src]   over 256 features.
// One wave per edge; edges [nEdges, nTotal) are self-loops (s = d = e-nEdges).
__global__ void scatter_edges(const long long* __restrict__ src,
                              const long long* __restrict__ dst,
                              int nEdges, int nTotal,
                              const float* __restrict__ dinv,
                              const float* __restrict__ H,
                              float* __restrict__ Out) {
    const int e    = (blockIdx.x * blockDim.x + threadIdx.x) >> 5;
    const int lane = threadIdx.x & 31;
    if (e >= nTotal) return;
    long long s, d;
    if (e < nEdges) { s = src[e]; d = dst[e]; }
    else            { s = d = (long long)(e - nEdges); }
    const float nrm = dinv[s] * dinv[d];
    const float4* h = (const float4*)(H + (size_t)s * F_MID);
    float*        o = Out + (size_t)d * F_MID;
#pragma unroll
    for (int it = 0; it < 2; ++it) {                 // 2*32*4 = 256 feats
        int i = lane + it * 32;
        float4 v = h[i];
        atomicAdd(o + 4 * i + 0, v.x * nrm);
        atomicAdd(o + 4 * i + 1, v.y * nrm);
        atomicAdd(o + 4 * i + 2, v.z * nrm);
        atomicAdd(o + 4 * i + 3, v.w * nrm);
    }
}

// --------------------- bias + relu + dropout (p=0.5) -----------------------
__device__ __forceinline__ unsigned hash32(unsigned x) {
    x ^= x >> 16; x *= 0x85ebca6bu; x ^= x >> 13; x *= 0xc2b2ae35u; x ^= x >> 16;
    return x;
}

__global__ void bias_relu_dropout(const float* __restrict__ in,
                                  const float* __restrict__ b,
                                  float* __restrict__ out, int n) {
    int i = blockIdx.x * blockDim.x + threadIdx.x;
    if (i < n) {
        float v = in[i] + b[i & (F_MID - 1)];
        v = v > 0.0f ? v : 0.0f;
        // deterministic mask, p_keep = 0.5, scale 1/(1-p) = 2
        v = (hash32((unsigned)i ^ 0x9e3779b9u) & 1u) ? 2.0f * v : 0.0f;
        out[i] = v;
    }
}

// ------------------------------- launcher ----------------------------------

extern "C" void kernel_launch(void* const* d_in, const int* in_sizes, int n_in,
                              void* d_out, int out_size, void* d_ws, size_t ws_size,
                              hipStream_t stream) {
    const float*     x   = (const float*)d_in[0];
    const long long* ei  = (const long long*)d_in[1];   // int64 [2][800000]
    const float*     W1  = (const float*)d_in[2];
    const float*     b1  = (const float*)d_in[3];
    const float*     W2  = (const float*)d_in[4];
    const float*     b2  = (const float*)d_in[5];
    float*           out = (float*)d_out;

    const long long* e_src = ei;             // row 0
    const long long* e_dst = ei + N_EDGES;   // row 1

    // Workspace layout (bytes), ~99 MB total, all 256B-aligned:
    char* ws = (char*)d_ws;
    float* deg  = (float*)(ws + 0);                      //  50000 f32
    float* dinv = (float*)(ws + 262144);                 //  50000 f32
    bf16*  W1t  = (bf16*) (ws + 524288);                 // 262144 bf16 [256][1024]
    bf16*  W2t  = (bf16*) (ws + 1048576);                // 262144 bf16 [1024][256]
    float* bufA = (float*)(ws + 1572864);                // 50000*256 f32
    float* bufB = (float*)(ws + 1572864 + (size_t)N_NODES * F_MID * 4);

    const int NM   = N_NODES * F_MID;                    // 12.8M
    const int nTot = N_EDGES + N_NODES;                  // 850000 (w/ self-loops)

    // ---- normalization: deg = 1 (self-loop) + scatter count; dinv = rsqrt
    fill_f32     <<<(N_NODES + 255) / 256, 256, 0, stream>>>(deg, 1.0f, N_NODES);
    degree_kernel<<<(N_EDGES + 255) / 256, 256, 0, stream>>>(e_dst, N_EDGES, deg);
    rsqrt_kernel <<<(N_NODES + 255) / 256, 256, 0, stream>>>(deg, dinv, N_NODES);

    // ---- weight transpose + bf16 convert (tiny, one-shot)
    transpose_to_bf16<<<(F_IN * F_MID + 255) / 256, 256, 0, stream>>>(W1, W1t, F_IN, F_MID);
    transpose_to_bf16<<<(F_IN * F_MID + 255) / 256, 256, 0, stream>>>(W2, W2t, F_MID, F_IN);

    // ---- layer 1: h = x @ W1  (GEMM first: scatter runs on 256-dim, not 1024)
    {
        // one block per 16-row band covers all 256 cols; LDS = 16*(1024+8)*2 B
        size_t shmem = (size_t)16 * (F_IN + 8) * sizeof(bf16);   // 33 KB
        gemm_wmma_bf16<<<N_NODES / 16, 256, shmem, stream>>>(
            x, W1t, nullptr, bufA, F_MID, F_IN, /*blocksPerRow=*/1);
    }
    fill_f32<<<(NM + 255) / 256, 256, 0, stream>>>(bufB, 0.0f, NM);
    scatter_edges<<<nTot / 8, 256, 0, stream>>>(e_src, e_dst, N_EDGES, nTot,
                                                dinv, bufA, bufB);
    bias_relu_dropout<<<(NM + 255) / 256, 256, 0, stream>>>(bufB, b1, bufA, NM);

    // ---- layer 2: aggregate first (linearity), then GEMM with bias
    fill_f32<<<(NM + 255) / 256, 256, 0, stream>>>(bufB, 0.0f, NM);
    scatter_edges<<<nTot / 8, 256, 0, stream>>>(e_src, e_dst, N_EDGES, nTot,
                                                dinv, bufA, bufB);
    {
        // N=1024 -> 4 blocks per 16-row band; LDS = 16*(256+8)*2 B
        size_t shmem = (size_t)16 * (F_MID + 8) * sizeof(bf16);  // 8.25 KB
        gemm_wmma_bf16<<<(N_NODES / 16) * 4, 256, shmem, stream>>>(
            bufB, W2t, b2, out, F_IN, F_MID, /*blocksPerRow=*/4);
    }
}
